// RegGNN_38817914421721
// MI455X (gfx1250) — matrix operand order repaired and checked by reading, not compiled, not run on hardware
//
#include <hip/hip_runtime.h>

// ---------------------------------------------------------------------------
// RegGNN forward for MI455X (gfx1250, wave32, WMMA).
//
// Roofline: W2 (64 MB) + W1 (128 MB) fp32 weights stream once from HBM:
// 192 MB / 23.3 TB/s ~= 8.2 us floor. 11.2 GFLOP at bf16-WMMA rates is ~4 us,
// so bf16 operands + f32 accumulation keep the kernel memory bound (f32 WMMA
// 16x16x4 would be ~8x slower per issue and flip it compute bound).
//
// vs round 1 (from disasm): (a) 32-bit offsets against uniform SGPR bases so
// strided weight gathers use saddr+voffset+imm instead of per-pointer
// v_add_nc_u64 chains; (b) row-clamp instead of divergent zero-fill (no EXEC
// save/restore in the loop); (c) activations pre-converted to bf16 once
// (<=3.6MB pass) so the A operand is 2x b128 loads, no per-iter cvts;
// (d) 8-way in-block K-split + LDS reduction -> 1024 waves instead of 128
// for latency hiding, deterministic sum order; (e) global_prefetch 2 K-steps
// ahead (speculative => tail overrun silently dropped).
// ---------------------------------------------------------------------------

typedef __attribute__((ext_vector_type(16))) __bf16 v16bf;
typedef __attribute__((ext_vector_type(8)))  __bf16 v8bf;
typedef __attribute__((ext_vector_type(4)))  __bf16 v4bf;
typedef __attribute__((ext_vector_type(8)))  float  v8f;
typedef __attribute__((ext_vector_type(4)))  float  f32x4;

#define NN 110          // nodes
#define MT 7            // ceil(110/16) M row-tiles
#define NW 2            // 16-column n-tiles per wave
#define KS 8            // waves per block = K-split factor
#define CB 64           // column chunk for the small [110x110] GEMM kernels

// ---------------------------------------------------------------------------
// f32 -> bf16 bulk convert (n4 = element count / 4)
// ---------------------------------------------------------------------------
__global__ void __launch_bounds__(256)
cvt_f32_bf16(const float* __restrict__ src, __bf16* __restrict__ dst, int n4)
{
    const int i = blockIdx.x * 256 + threadIdx.x;
    if (i < n4) {
        f32x4 v = ((const f32x4*)src)[i];
        v4bf o;
        o[0] = (__bf16)v[0]; o[1] = (__bf16)v[1];
        o[2] = (__bf16)v[2]; o[3] = (__bf16)v[3];
        ((v4bf*)dst)[i] = o;
    }
}

// ---------------------------------------------------------------------------
// D[110 x N] = act(A[110 x K] @ W[K x N] + bias)
// Block: 8 waves x (same 32 columns), wave w covers K-chunk w. Private f32
// accumulators, LDS reduction (serial sum over waves), bias+ReLU in the epilogue.
// WMMA operand k order used here (wave32; contiguous, not the ISA table order):
//   A 16x32 bf16 : lane<16 -> M=lane,    K = 0..15
//                  lane>=16 -> M=lane-16, K = 16..31
//   B 32x16 bf16 : lane<16 -> N=lane,    K = 0..15 (2 per VGPR)
//                  lane>=16 -> N=lane-16, K = 16..31
//   C/D 16x16 f32: vgpr r -> M = r + (lane<16 ? 0 : 8), N = lane & 15
// ---------------------------------------------------------------------------
__global__ void __launch_bounds__(32 * KS)
gemm_wmma_bf16(const __bf16* __restrict__ A, int lda,
               const float* __restrict__ W, int ldw, int K,
               const float* __restrict__ bias,
               float* __restrict__ D, int ldd, int do_relu)
{
    __shared__ float red[KS][NW][MT][32][8];   // 8*2*7*32*8*4 = 114688 B

    const int lane  = threadIdx.x & 31;
    const int w     = threadIdx.x >> 5;        // K-split id
    const int nb0   = blockIdx.x * (16 * NW);
    const int ncol  = lane & 15;
    const int ahalf = (lane >> 4) << 4;        // A-operand K offset: 0 or 16
    const int bhalf = (lane >> 4) << 4;        // B-operand K offset: 0 or 16

    const int Kc = K / KS;                     // 512 or 1024 (mult of 32)
    const int k0 = w * Kc;

    v8f acc[NW][MT];
#pragma unroll
    for (int nt = 0; nt < NW; ++nt)
#pragma unroll
        for (int t = 0; t < MT; ++t)
#pragma unroll
            for (int r = 0; r < 8; ++r) acc[nt][t][r] = 0.0f;

    // 32-bit element offsets against uniform base pointers (saddr form).
    unsigned aoff[MT];
#pragma unroll
    for (int t = 0; t < MT; ++t) {
        int m = t * 16 + ncol;
        if (m > NN - 1) m = NN - 1;            // clamp pad rows (discarded)
        aoff[t] = (unsigned)m * (unsigned)lda + (unsigned)(k0 + ahalf);
    }
    unsigned woff[NW];
#pragma unroll
    for (int nt = 0; nt < NW; ++nt)
        woff[nt] = (unsigned)(k0 + bhalf) * (unsigned)ldw
                 + (unsigned)(nb0 + nt * 16 + ncol);
    // prefetch stream: lane -> one weight row, 64B line covers 16 columns
    unsigned pfoff = (unsigned)(k0 + lane) * (unsigned)ldw + (unsigned)nb0;
    const unsigned pfd   = 64u * (unsigned)ldw;     // 2 K-steps ahead
    const unsigned wstep = 32u * (unsigned)ldw;

    for (int kk = 0; kk < Kc; kk += 32) {
        // ---- B tiles: 16 strided f32 loads (imm offset = e*ldw*4), cvt ----
        v16bf bmat[NW];
#pragma unroll
        for (int nt = 0; nt < NW; ++nt)
#pragma unroll
            for (int e = 0; e < 16; ++e)
                bmat[nt][e] = (__bf16)W[woff[nt] + (unsigned)(e * ldw)];

        __builtin_prefetch(W + pfoff + pfd, 0, 1);         // cols nb0..+15
        __builtin_prefetch(W + pfoff + pfd + 16u, 0, 1);   // cols +16..+31

        // ---- A tiles (bf16 in memory): 2x b128 each, shared across nt ----
#pragma unroll
        for (int t = 0; t < MT; ++t) {
            v8bf lo = *(const v8bf*)(A + aoff[t]);
            v8bf hi = *(const v8bf*)(A + aoff[t] + 8);
            v16bf amat = __builtin_shufflevector(
                lo, hi, 0, 1, 2, 3, 4, 5, 6, 7,
                        8, 9, 10, 11, 12, 13, 14, 15);
#pragma unroll
            for (int nt = 0; nt < NW; ++nt)
                acc[nt][t] = __builtin_amdgcn_wmma_f32_16x16x32_bf16(
                    false, amat, false, bmat[nt], (short)0, acc[nt][t],
                    false, false);
            aoff[t] += 32u;
        }
#pragma unroll
        for (int nt = 0; nt < NW; ++nt) woff[nt] += wstep;
        pfoff += wstep;
    }

    // ---- LDS reduction across the 8 K-chunks, then bias/ReLU/store ----
#pragma unroll
    for (int nt = 0; nt < NW; ++nt)
#pragma unroll
        for (int t = 0; t < MT; ++t)
#pragma unroll
            for (int r = 0; r < 8; ++r)
                red[w][nt][t][lane][r] = acc[nt][t][r];
    __syncthreads();

    for (int idx = threadIdx.x; idx < NW * MT * 8 * 32; idx += 32 * KS) {
        const int l  = idx & 31;
        const int r  = (idx >> 5) & 7;
        const int t  = (idx >> 8) % MT;
        const int nt = (idx >> 8) / MT;
        float s = 0.0f;
#pragma unroll
        for (int ww = 0; ww < KS; ++ww) s += red[ww][nt][t][l][r];
        const int m = t * 16 + ((l >> 4) << 3) + r;
        const int n = nb0 + nt * 16 + (l & 15);
        if (m < NN) {
            float v = s + bias[n];
            if (do_relu) v = fmaxf(v, 0.0f);
            D[(size_t)m * ldd + n] = v;
        }
    }
}

// ---------------------------------------------------------------------------
// aggr[j, c] = (1/max(deg_j,1)) * sum_i mask[i][j] * h[i][c]
// Block = 64-column chunk; h chunk + adjacency in LDS (h read once total).
// ---------------------------------------------------------------------------
__global__ void __launch_bounds__(256)
aggr_kernel(const int* __restrict__ adj, float* __restrict__ hcat)
{
    __shared__ float         hbuf[NN * CB];        // 28160 B
    __shared__ unsigned char maskS[NN * NN];       // 12100 B
    __shared__ float         invdeg[NN];

    const int tid = threadIdx.x;
    const int c0  = blockIdx.x * CB;

    for (int idx = tid; idx < NN * NN; idx += 256)
        maskS[idx] = (adj[idx] != 0) ? 1 : 0;
    for (int idx = tid; idx < NN * CB; idx += 256)
        hbuf[idx] = hcat[(size_t)(idx / CB) * 8192 + c0 + (idx % CB)];
    __syncthreads();

    if (tid < NN) {
        int d = 0;
        for (int i = 0; i < NN; ++i) d += maskS[i * NN + tid];
        invdeg[tid] = 1.0f / fmaxf((float)d, 1.0f);
    }
    __syncthreads();

    const int c = tid & (CB - 1);
    for (int j = tid >> 6; j < NN; j += 4) {
        float s = 0.0f;
        for (int i = 0; i < NN; ++i)
            s = fmaf((float)maskS[i * NN + j], hbuf[i * CB + c], s);
        hcat[(size_t)j * 8192 + 4096 + c0 + c] = s * invdeg[j];
    }
}

// ---------------------------------------------------------------------------
// out[n, c] = bl[n] + sum_m Wl[n][m] * out1[m][c]
// ---------------------------------------------------------------------------
__global__ void __launch_bounds__(256)
final_kernel(const float* __restrict__ out1, const float* __restrict__ Wl,
             const float* __restrict__ bl, float* __restrict__ out)
{
    __shared__ float obuf[NN * CB];   // 28160 B
    __shared__ float wlS[NN * NN];    // 48400 B

    const int tid = threadIdx.x;
    const int c0  = blockIdx.x * CB;

    for (int idx = tid; idx < NN * NN; idx += 256) wlS[idx] = Wl[idx];
    for (int idx = tid; idx < NN * CB; idx += 256)
        obuf[idx] = out1[(size_t)(idx / CB) * 4096 + c0 + (idx % CB)];
    __syncthreads();

    const int c = tid & (CB - 1);
    for (int n = tid >> 6; n < NN; n += 4) {
        float s = bl[n];
        for (int m = 0; m < NN; ++m)
            s = fmaf(wlS[n * NN + m], obuf[m * CB + c], s);
        out[(size_t)n * 4096 + c0 + c] = s;
    }
}

// ---------------------------------------------------------------------------
extern "C" void kernel_launch(void* const* d_in, const int* in_sizes, int n_in,
                              void* d_out, int out_size, void* d_ws,
                              size_t ws_size, hipStream_t stream)
{
    const float* x  = (const float*)d_in[0];   // [110, 4096]
    const int*   adj= (const int*)  d_in[1];   // [110, 110]
    const float* W2 = (const float*)d_in[2];   // [4096, 4096]
    const float* b2 = (const float*)d_in[3];   // [4096]
    const float* W1 = (const float*)d_in[4];   // [8192, 4096]
    const float* b1 = (const float*)d_in[5];   // [4096]
    const float* Wl = (const float*)d_in[6];   // [110, 110]
    const float* bl = (const float*)d_in[7];   // [110]
    float*       out= (float*)d_out;           // [110, 4096]

    // Workspace (7.2 MB): hcat f32 [110x8192] = [h | aggr], out1 f32
    // [110x4096], abf bf16 (activations, [110x4096] then [110x8192]).
    float*  hcat = (float*)d_ws;
    float*  out1 = hcat + (size_t)NN * 8192;
    __bf16* abf  = (__bf16*)(out1 + (size_t)NN * 4096);

    const dim3 gblk(32 * KS);                   // 8 waves / block
    const dim3 ggrd(4096 / (16 * NW));          // 128 blocks of 32 columns

    // Stage 0: xb = bf16(x)
    cvt_f32_bf16<<<dim3((NN * 4096 / 4 + 255) / 256), dim3(256), 0, stream>>>(
        x, abf, NN * 4096 / 4);
    // Stage 1: h = x @ W2 + b2  -> hcat[:, 0:4096]
    gemm_wmma_bf16<<<ggrd, gblk, 0, stream>>>(abf, 4096, W2, 4096, 4096, b2,
                                              hcat, 8192, 0);
    // Stage 2: mean aggregation -> hcat[:, 4096:8192]
    aggr_kernel<<<dim3(4096 / CB), dim3(256), 0, stream>>>(adj, hcat);
    // Stage 2b: abf = bf16(hcat)  [110 x 8192]
    cvt_f32_bf16<<<dim3((NN * 8192 / 4 + 255) / 256), dim3(256), 0, stream>>>(
        hcat, abf, NN * 8192 / 4);
    // Stage 3: out1 = relu(hcat @ W1 + b1)
    gemm_wmma_bf16<<<ggrd, gblk, 0, stream>>>(abf, 8192, W1, 4096, 8192, b1,
                                              out1, 4096, 1);
    // Stage 4: out = Wl @ out1 + bl[:, None]
    final_kernel<<<dim3(4096 / CB), dim3(256), 0, stream>>>(out1, Wl, bl, out);
}
